// WindowAttention1D_56564719288662
// MI455X (gfx1250) — compile-verified, matrix operand order checked by reading, not run on hardware
//
#include <hip/hip_runtime.h>

typedef __attribute__((ext_vector_type(16))) __bf16 v16bf;
typedef __attribute__((ext_vector_type(8)))  float  v8f;

#define C_DIM 768
#define HEADS 12
#define HD    64
#define SEQ   128
#define NWIN  64
#define BTOT  256             /* B_ = NW * B */
#define MTOT  (BTOT * SEQ)    /* 32768 rows  */
#define NQKV  (3 * C_DIM)     /* 2304 cols   */

static __device__ __forceinline__ v8f wmma_bf16(v16bf a, v16bf b, v8f c) {
  // 8 args: (neg_a, A, neg_b, B, c_mod, C, reuse_a, reuse_b)
  return __builtin_amdgcn_wmma_f32_16x16x32_bf16(false, a, false, b, (short)0, c,
                                                 false, false);
}

static __device__ __forceinline__ v16bf cvt16_f32_bf16(const float* p) {
  v16bf r;
#pragma unroll
  for (int i = 0; i < 16; i += 4) {
    float4 f = *(const float4*)(p + i);
    r[i + 0] = (__bf16)f.x; r[i + 1] = (__bf16)f.y;
    r[i + 2] = (__bf16)f.z; r[i + 3] = (__bf16)f.w;
  }
  return r;
}

// ---------------------------------------------------------------------------
// Elementwise fp32 -> bf16 conversion (weights only; x converts in-GEMM)
// ---------------------------------------------------------------------------
__global__ void cvt_f32_bf16_kernel(const float* __restrict__ in,
                                    __bf16* __restrict__ out, int n) {
  int i = blockIdx.x * blockDim.x + threadIdx.x;
  int stride = gridDim.x * blockDim.x;
  for (; i < n; i += stride) out[i] = (__bf16)in[i];
}

// ---------------------------------------------------------------------------
// QKV projection: qkv = x @ qkv_w^T   (A 32768x768 fp32, B 2304x768 bf16, NT)
// Block = 256 thr = 8 waves (4 in M x 2 in N); wave tile 32x64; block 128x128.
// 8 independent accumulators per wave hide the WMMA RAW hazard window.
// Epilogue scatters to Q (scaled), K, and V^T bf16 buffers.
// ---------------------------------------------------------------------------
__global__ __launch_bounds__(256)
void qkv_gemm_kernel(const float* __restrict__ x, const __bf16* __restrict__ wb,
                     __bf16* __restrict__ qb, __bf16* __restrict__ kb,
                     __bf16* __restrict__ vtb) {
  const int lane  = threadIdx.x & 31;
  const int wid   = threadIdx.x >> 5;
  const int waveM = wid & 3, waveN = wid >> 2;
  const int rowBase = blockIdx.x * 128 + waveM * 32;
  const int colBase = blockIdx.y * 128 + waveN * 64;
  const int lm = lane & 15;
  const int lk = (lane >> 4) * 16;

  v8f acc[2][4] = {};

  const float* ap[2];
  const __bf16* bp[4];
#pragma unroll
  for (int i = 0; i < 2; ++i)
    ap[i] = x + (size_t)(rowBase + i * 16 + lm) * C_DIM;
#pragma unroll
  for (int i = 0; i < 4; ++i)
    bp[i] = wb + (size_t)(colBase + i * 16 + lm) * C_DIM;

#pragma unroll 2
  for (int k0 = 0; k0 < C_DIM; k0 += 32) {
    v16bf af[2], bf[4];
#pragma unroll
    for (int i = 0; i < 2; ++i) af[i] = cvt16_f32_bf16(ap[i] + k0 + lk);
#pragma unroll
    for (int i = 0; i < 4; ++i) bf[i] = *(const v16bf*)(bp[i] + k0 + lk);
#pragma unroll
    for (int tm = 0; tm < 2; ++tm)
#pragma unroll
      for (int tn = 0; tn < 4; ++tn)
        acc[tm][tn] = wmma_bf16(af[tm], bf[tn], acc[tm][tn]);
  }

  // Scatter epilogue. Column tiles are 16-wide and 16-aligned, so `which`
  // (q/k/v) and head are uniform within a tile.
#pragma unroll
  for (int tm = 0; tm < 2; ++tm) {
#pragma unroll
    for (int tn = 0; tn < 4; ++tn) {
      const int colT  = colBase + tn * 16;
      const int which = colT / C_DIM;         // 0=q 1=k 2=v
      const int rem   = colT % C_DIM;
      const int hh    = rem / HD;
      const int d     = (rem % HD) + lm;
#pragma unroll
      for (int r = 0; r < 8; ++r) {
        const int row = rowBase + tm * 16 + r + ((lane >> 4) << 3);
        const int bb  = row >> 7;             // /SEQ
        const int n   = row & (SEQ - 1);
        const float v = acc[tm][tn][r];
        if (which == 0) {
          qb[(((size_t)bb * HEADS + hh) * SEQ + n) * HD + d] = (__bf16)(v * 0.125f);
        } else if (which == 1) {
          kb[(((size_t)bb * HEADS + hh) * SEQ + n) * HD + d] = (__bf16)v;
        } else { // V stored transposed (hd, seq) so P@V B-operand is contiguous
          vtb[(((size_t)bb * HEADS + hh) * HD + d) * SEQ + n] = (__bf16)v;
        }
      }
    }
  }
}

// ---------------------------------------------------------------------------
// Attention: one block per (batch, head). 8 waves; wave wv owns 16 query rows.
// S = Q K^T (WMMA) + rel-pos bias + window mask -> softmax (register resident,
// half-wave shfl_xor reductions) -> P through LDS -> O = P V (WMMA).
// ---------------------------------------------------------------------------
__global__ __launch_bounds__(256)
void attn_kernel(const float* __restrict__ mask, const float* __restrict__ bias_table,
                 const __bf16* __restrict__ qb, const __bf16* __restrict__ kb,
                 const __bf16* __restrict__ vtb, __bf16* __restrict__ ob) {
  __shared__ __bf16 plds[8 * 16 * SEQ] __attribute__((aligned(32)));

  const int bh   = blockIdx.x;
  const int b    = bh / HEADS;
  const int h    = bh % HEADS;
  const int w    = b & (NWIN - 1);
  const int lane = threadIdx.x & 31;
  const int wv   = threadIdx.x >> 5;
  const int lm   = lane & 15;
  const int lhalf = lane >> 4;
  const int lk   = lhalf * 16;

  const __bf16* Q  = qb  + (size_t)bh * SEQ * HD;
  const __bf16* K  = kb  + (size_t)bh * SEQ * HD;
  const __bf16* VT = vtb + (size_t)bh * HD * SEQ;

  // ---- S = Q K^T : 16 rows x 128 cols, K-dim = 64 (two 32-steps) ----
  v8f s[8] = {};
  v16bf aq0 = *(const v16bf*)(Q + (wv * 16 + lm) * HD +  0 + lk);
  v16bf aq1 = *(const v16bf*)(Q + (wv * 16 + lm) * HD + 32 + lk);
#pragma unroll
  for (int t = 0; t < 8; ++t) {
    v16bf bk0 = *(const v16bf*)(K + (t * 16 + lm) * HD +  0 + lk);
    v16bf bk1 = *(const v16bf*)(K + (t * 16 + lm) * HD + 32 + lk);
    s[t] = wmma_bf16(aq0, bk0, s[t]);
    s[t] = wmma_bf16(aq1, bk1, s[t]);
  }

  // ---- relative position bias + window mask ----
#pragma unroll
  for (int t = 0; t < 8; ++t) {
    const int j = t * 16 + lm;
#pragma unroll
    for (int r = 0; r < 8; ++r) {
      const int i = wv * 16 + r + (lhalf << 3);
      const float bv = bias_table[(i - j + SEQ - 1) * HEADS + h];
      const float mv = mask[((size_t)w * SEQ + i) * SEQ + j];
      s[t][r] += bv + mv;
    }
  }

  // ---- softmax: rows m live on lanes 0-15, rows m+8 on lanes 16-31 ----
  float inv[8];
#pragma unroll
  for (int r = 0; r < 8; ++r) {
    float m = s[0][r];
#pragma unroll
    for (int t = 1; t < 8; ++t) m = fmaxf(m, s[t][r]);
    m = fmaxf(m, __shfl_xor(m, 1, 32));
    m = fmaxf(m, __shfl_xor(m, 2, 32));
    m = fmaxf(m, __shfl_xor(m, 4, 32));
    m = fmaxf(m, __shfl_xor(m, 8, 32));
    float sum = 0.0f;
#pragma unroll
    for (int t = 0; t < 8; ++t) {
      const float e = __expf(s[t][r] - m);
      s[t][r] = e;
      sum += e;
    }
    sum += __shfl_xor(sum, 1, 32);
    sum += __shfl_xor(sum, 2, 32);
    sum += __shfl_xor(sum, 4, 32);
    sum += __shfl_xor(sum, 8, 32);
    inv[r] = 1.0f / sum;
  }

  // ---- P -> LDS (bf16, row-major 16x128 per wave) for A-fragment layout ----
  __bf16* pw = plds + wv * 16 * SEQ;
#pragma unroll
  for (int t = 0; t < 8; ++t)
#pragma unroll
    for (int r = 0; r < 8; ++r)
      pw[(r + (lhalf << 3)) * SEQ + t * 16 + lm] = (__bf16)(s[t][r] * inv[r]);
  __syncthreads();

  // ---- O = P @ V : 16 x 64, K-dim = 128 (four 32-steps) ----
  v8f o[4] = {};
#pragma unroll
  for (int ks = 0; ks < 4; ++ks) {
    v16bf ap = *(const v16bf*)(pw + lm * SEQ + ks * 32 + lk);
#pragma unroll
    for (int tv = 0; tv < 4; ++tv) {
      v16bf bv = *(const v16bf*)(VT + (tv * 16 + lm) * SEQ + ks * 32 + lk);
      o[tv] = wmma_bf16(ap, bv, o[tv]);
    }
  }

  // ---- write O as (B_, N, C) bf16, ready for proj GEMM ----
#pragma unroll
  for (int tv = 0; tv < 4; ++tv)
#pragma unroll
    for (int r = 0; r < 8; ++r) {
      const int i = wv * 16 + r + (lhalf << 3);
      const int d = tv * 16 + lm;
      ob[((size_t)b * SEQ + i) * C_DIM + h * HD + d] = (__bf16)o[tv][r];
    }
}

// ---------------------------------------------------------------------------
// Output projection: out = O @ proj_w^T + proj_b  (fp32 out)
// Same 128x128 block tile / 32x64 wave tile as the QKV GEMM.
// ---------------------------------------------------------------------------
__global__ __launch_bounds__(256)
void proj_gemm_kernel(const __bf16* __restrict__ A, const __bf16* __restrict__ wb,
                      const float* __restrict__ bias, float* __restrict__ out) {
  const int lane  = threadIdx.x & 31;
  const int wid   = threadIdx.x >> 5;
  const int waveM = wid & 3, waveN = wid >> 2;
  const int rowBase = blockIdx.x * 128 + waveM * 32;
  const int colBase = blockIdx.y * 128 + waveN * 64;
  const int lm = lane & 15;
  const int lk = (lane >> 4) * 16;

  v8f acc[2][4] = {};

  const __bf16* ap[2];
  const __bf16* bp[4];
#pragma unroll
  for (int i = 0; i < 2; ++i)
    ap[i] = A + (size_t)(rowBase + i * 16 + lm) * C_DIM;
#pragma unroll
  for (int i = 0; i < 4; ++i)
    bp[i] = wb + (size_t)(colBase + i * 16 + lm) * C_DIM;

#pragma unroll 2
  for (int k0 = 0; k0 < C_DIM; k0 += 32) {
    v16bf af[2], bf[4];
#pragma unroll
    for (int i = 0; i < 2; ++i) af[i] = *(const v16bf*)(ap[i] + k0 + lk);
#pragma unroll
    for (int i = 0; i < 4; ++i) bf[i] = *(const v16bf*)(bp[i] + k0 + lk);
#pragma unroll
    for (int tm = 0; tm < 2; ++tm)
#pragma unroll
      for (int tn = 0; tn < 4; ++tn)
        acc[tm][tn] = wmma_bf16(af[tm], bf[tn], acc[tm][tn]);
  }

#pragma unroll
  for (int tm = 0; tm < 2; ++tm)
#pragma unroll
    for (int tn = 0; tn < 4; ++tn) {
      const int col = colBase + tn * 16 + lm;
      const float pb = bias[col];
#pragma unroll
      for (int r = 0; r < 8; ++r) {
        const int row = rowBase + tm * 16 + r + ((lane >> 4) << 3);
        out[(size_t)row * C_DIM + col] = acc[tm][tn][r] + pb;
      }
    }
}

// ---------------------------------------------------------------------------
extern "C" void kernel_launch(void* const* d_in, const int* in_sizes, int n_in,
                              void* d_out, int out_size, void* d_ws, size_t ws_size,
                              hipStream_t stream) {
  (void)in_sizes; (void)n_in; (void)out_size; (void)ws_size;
  const float* x          = (const float*)d_in[0];
  const float* mask       = (const float*)d_in[1];
  const float* qkv_w      = (const float*)d_in[2];
  const float* bias_table = (const float*)d_in[3];
  const float* proj_w     = (const float*)d_in[4];
  const float* proj_b     = (const float*)d_in[5];

  char* ws = (char*)d_ws;
  size_t off = 0;
  auto carve = [&](size_t bytes) -> void* {
    void* p = ws + off;
    off += (bytes + 255) & ~(size_t)255;
    return p;
  };
  __bf16* wqkvb  = (__bf16*)carve((size_t)NQKV * C_DIM * sizeof(__bf16));
  __bf16* wprojb = (__bf16*)carve((size_t)C_DIM * C_DIM * sizeof(__bf16));
  __bf16* qb     = (__bf16*)carve((size_t)BTOT * HEADS * SEQ * HD * sizeof(__bf16));
  __bf16* kb     = (__bf16*)carve((size_t)BTOT * HEADS * SEQ * HD * sizeof(__bf16));
  __bf16* vtb    = (__bf16*)carve((size_t)BTOT * HEADS * HD * SEQ * sizeof(__bf16));
  __bf16* ob     = (__bf16*)carve((size_t)MTOT * C_DIM * sizeof(__bf16));

  const int n1 = NQKV * C_DIM;
  cvt_f32_bf16_kernel<<<(n1 + 255) / 256, 256, 0, stream>>>(qkv_w, wqkvb, n1);
  const int n2 = C_DIM * C_DIM;
  cvt_f32_bf16_kernel<<<(n2 + 255) / 256, 256, 0, stream>>>(proj_w, wprojb, n2);

  qkv_gemm_kernel<<<dim3(MTOT / 128, NQKV / 128), 256, 0, stream>>>(x, wqkvb, qb, kb, vtb);
  attn_kernel<<<BTOT * HEADS, 256, 0, stream>>>(mask, bias_table, qb, kb, vtb, ob);
  proj_gemm_kernel<<<dim3(MTOT / 128, C_DIM / 128), 256, 0, stream>>>(ob, wprojb, proj_b,
                                                                      (float*)d_out);
}
